// NicheST_10780367913474
// MI455X (gfx1250) — matrix operand order repaired and checked by reading, not compiled, not run on hardware
//
#include <hip/hip_runtime.h>

// ---------------- types ----------------
typedef __attribute__((ext_vector_type(16))) __bf16 v16bf;
typedef __attribute__((ext_vector_type(8)))  __bf16 v8bf;
typedef __attribute__((ext_vector_type(8)))  float  v8f;
typedef __attribute__((ext_vector_type(4)))  unsigned int v4u;
typedef __attribute__((ext_vector_type(8)))  int v8i;
typedef __attribute__((ext_vector_type(4)))  int v4i;

#define NN      4096
#define IN_DIM  3000
#define INP     3008            /* IN_DIM padded to 32 (K) / 16 (N) */
#define H1DIM   512
#define DDIM    64
#define KSUB    32
#define NE      49152
#define NHEADS  4
#define NNP     (NN + 16)       /* V^T row stride, padded */

#if __has_builtin(__builtin_amdgcn_tensor_load_to_lds)
#define HAVE_TDM 1
#else
#define HAVE_TDM 0
#endif

static __device__ __forceinline__ v8f wmma_bf16(v16bf a, v16bf b, v8f c) {
  return __builtin_amdgcn_wmma_f32_16x16x32_bf16(false, a, false, b, (short)0, c, false, false);
}

// A-fragment: lane holds one row; K contiguous {h*16..h*16+15}
static __device__ __forceinline__ v16bf load_frag_split(const __bf16* p) {
  const v8bf x0 = *(const v8bf*)(p);
  const v8bf x1 = *(const v8bf*)(p + 8);
  return __builtin_shufflevector(x0, x1, 0, 1, 2, 3, 4, 5, 6, 7,
                                 8, 9, 10, 11, 12, 13, 14, 15);
}

// ---------------- utility kernels ----------------
__global__ void zero_k(float* __restrict__ p, size_t n) {
  size_t i  = (size_t)blockIdx.x * blockDim.x + threadIdx.x;
  size_t st = (size_t)gridDim.x * blockDim.x;
  for (; i < n; i += st) p[i] = 0.0f;
}

__global__ void deg_acc_k(const int* __restrict__ dst, float* __restrict__ deg, int E) {
  int i  = blockIdx.x * blockDim.x + threadIdx.x;
  int st = gridDim.x * blockDim.x;
  for (; i < E; i += st) atomicAdd(&deg[dst[i]], 1.0f);
}

__global__ void inv_sqrt_k(float* __restrict__ deg, int n) {
  int i = blockIdx.x * blockDim.x + threadIdx.x;
  if (i < n) deg[i] = rsqrtf(deg[i] + 1.0f);
}

// ---------------- bf16 packing ----------------
// Ap[M][Kp] = bf16(A[M][K]), zero padded in K.
__global__ void pack_a_k(const float* __restrict__ A, __bf16* __restrict__ Ap,
                         int M, int K, int Kp) {
  size_t idx = (size_t)blockIdx.x * blockDim.x + threadIdx.x;
  const size_t tot = (size_t)M * Kp / 8;
  const size_t st  = (size_t)gridDim.x * blockDim.x;
  for (; idx < tot; idx += st) {
    const size_t o  = idx * 8;
    const int    mr = (int)(o / Kp);
    const int    k0 = (int)(o % Kp);
    v8bf v;
    if (k0 + 8 <= K) {
      const float4 f0 = *(const float4*)(A + (size_t)mr * K + k0);
      const float4 f1 = *(const float4*)(A + (size_t)mr * K + k0 + 4);
      v[0] = (__bf16)f0.x; v[1] = (__bf16)f0.y; v[2] = (__bf16)f0.z; v[3] = (__bf16)f0.w;
      v[4] = (__bf16)f1.x; v[5] = (__bf16)f1.y; v[6] = (__bf16)f1.z; v[7] = (__bf16)f1.w;
    } else {
      #pragma unroll
      for (int i = 0; i < 8; ++i) {
        const int k = k0 + i;
        v[i] = (k < K) ? (__bf16)A[(size_t)mr * K + k] : (__bf16)0.0f;
      }
    }
    *(v8bf*)(Ap + o) = v;
  }
}

// Bt[Np][Kp] = bf16(op(B)), zero padded. trans=1: B is [K,N]; trans=0: B is [N,K].
__global__ void pack_bt_k(const float* __restrict__ B, __bf16* __restrict__ Bt,
                          int K, int N, int Np, int Kp, int trans) {
  size_t idx = (size_t)blockIdx.x * blockDim.x + threadIdx.x;
  const size_t tot = (size_t)Np * Kp / 8;
  const size_t st  = (size_t)gridDim.x * blockDim.x;
  for (; idx < tot; idx += st) {
    const size_t o  = idx * 8;
    const int    n  = (int)(o / Kp);
    const int    k0 = (int)(o % Kp);
    v8bf v;
    #pragma unroll
    for (int i = 0; i < 8; ++i) {
      const int k = k0 + i;
      float f = 0.0f;
      if (n < N && k < K) f = trans ? B[(size_t)k * N + n] : B[(size_t)n * K + k];
      v[i] = (__bf16)f;
    }
    *(v8bf*)(Bt + o) = v;
  }
}

// ---------------- LDS-free bf16 WMMA GEMM ----------------
// C[M,N] = act(alpha * Ap[M,Kp] @ Bt[Np,Kp]^T + bias); one wave per 16x16 tile.
__global__ __launch_bounds__(32)
void gemm_bf16_k(const __bf16* __restrict__ Ap, const __bf16* __restrict__ Bt,
                 const float* __restrict__ bias, float* __restrict__ C,
                 int M, int N, int Kp, int relu, float alpha) {
  const int lane = threadIdx.x;
  const int r    = lane & 15;
  const int hi   = lane >> 4;
  const int n0   = blockIdx.x * 16;
  const int m0   = blockIdx.y * 16;

  const __bf16* a_ptr = Ap + (size_t)(m0 + r) * Kp + hi * 16;  // contiguous frag
  const __bf16* b_ptr = Bt + (size_t)(n0 + r) * Kp + hi * 16;  // contiguous frag

  v8f acc = {};
  int k0 = 0;
  for (; k0 + 64 <= Kp; k0 += 64) {
    __builtin_prefetch(a_ptr + k0 + 256, 0, 1);
    __builtin_prefetch(b_ptr + k0 + 256, 0, 1);
    const v16bf a0 = load_frag_split(a_ptr + k0);
    const v16bf b0 = *(const v16bf*)(b_ptr + k0);
    const v16bf a1 = load_frag_split(a_ptr + k0 + 32);
    const v16bf b1 = *(const v16bf*)(b_ptr + k0 + 32);
    acc = wmma_bf16(a0, b0, acc);
    acc = wmma_bf16(a1, b1, acc);
  }
  for (; k0 < Kp; k0 += 32) {
    const v16bf a = load_frag_split(a_ptr + k0);
    const v16bf b = *(const v16bf*)(b_ptr + k0);
    acc = wmma_bf16(a, b, acc);
  }

  const int col = n0 + r;
  if (col < N) {
    const float bb = bias ? bias[col] : 0.0f;
    #pragma unroll
    for (int i = 0; i < 8; ++i) {
      const int row = m0 + i + hi * 8;
      float v = acc[i] * alpha + bb;
      if (relu) v = fmaxf(v, 0.0f);
      C[(size_t)row * N + col] = v;
    }
  }
}

// ---------------- GCN edge aggregation + finalize ----------------
__global__ void edge_agg_k(const int* __restrict__ src, const int* __restrict__ dst,
                           const float* __restrict__ h, const float* __restrict__ inv,
                           float* __restrict__ agg, int E, int F) {
  const int e = blockIdx.x;
  if (e >= E) return;
  const int s = src[e], d = dst[e];
  const float c = inv[s] * inv[d];
  for (int f = threadIdx.x; f < F; f += blockDim.x)
    atomicAdd(&agg[(size_t)d * F + f], h[(size_t)s * F + f] * c);
}

__global__ void gcn_finalize_k(float* __restrict__ agg, const float* __restrict__ h,
                               const float* __restrict__ inv, const float* __restrict__ b,
                               int F, int relu) {
  const int n   = blockIdx.x;
  const float s = inv[n] * inv[n];
  for (int f = threadIdx.x; f < F; f += blockDim.x) {
    float v = agg[(size_t)n * F + f] + h[(size_t)n * F + f] * s + b[f];
    if (relu) v = fmaxf(v, 0.0f);
    agg[(size_t)n * F + f] = v;
  }
}

// ---------------- subgraph gather + mean pool ----------------
__global__ void gather_mean_k(const float* __restrict__ spot, const int* __restrict__ sub,
                              float* __restrict__ loc) {
  const int n = blockIdx.x;
  const int d = threadIdx.x;   // 64
  float s = 0.0f;
  #pragma unroll 4
  for (int k = 0; k < KSUB; ++k)
    s += spot[(size_t)sub[n * KSUB + k] * DDIM + d];
  loc[(size_t)n * DDIM + d] = s * (1.0f / (float)KSUB);
}

// ---------------- LOC -> bf16 pack, staged through LDS by the Tensor Data Mover ----
// One 64x64 fp32 tile per block (16 KB). Wave 0 issues the TDM descriptor, waits on
// TENSORcnt, then all 8 waves convert LDS fp32 -> global bf16.
__global__ __launch_bounds__(256)
void pack_loc_tdm_k(const float* __restrict__ LOC, __bf16* __restrict__ LOCp) {
  __shared__ __attribute__((aligned(16))) float tile[64 * DDIM];
  const size_t base = (size_t)blockIdx.x * 64 * DDIM;
#if HAVE_TDM
  if (threadIdx.x < 32) {
    const unsigned long long ga = (unsigned long long)(const void*)(LOC + base);
    const unsigned int lds = (unsigned int)(unsigned long long)(void*)&tile[0];
    v4u g0; v8i g1; v4i g2, g3;
    g0[0] = 1u;                                   // count=1 (valid descriptor)
    g0[1] = lds;                                  // lds_addr
    g0[2] = (unsigned int)ga;                     // global_addr[31:0]
    g0[3] = (unsigned int)(ga >> 32) | (2u << 30);// global_addr[56:32] | type=2
    g1[0] = (int)(2u << 16);                      // data_size=4B
    g1[1] = (int)((DDIM & 0xffffu) << 16);        // tensor_dim0 lo
    g1[2] = (int)((NN & 0xffffu) << 16);          // tensor_dim0 hi | tensor_dim1 lo
    g1[3] = (int)((64u & 0xffffu) << 16);         // tensor_dim1 hi | tile_dim0=64
    g1[4] = 64;                                   // tile_dim1=64, tile_dim2=0
    g1[5] = DDIM;                                 // tensor_dim0_stride lo
    g1[6] = 0;                                    // stride0 hi | stride1 lo
    g1[7] = 0;                                    // stride1 hi
    g2 = (v4i){0, 0, 0, 0};
    g3 = (v4i){0, 0, 0, 0};
#if __clang_major__ >= 23
    v8i g4 = (v8i){0, 0, 0, 0, 0, 0, 0, 0};
    __builtin_amdgcn_tensor_load_to_lds(g0, g1, g2, g3, g4, 0);
#else
    __builtin_amdgcn_tensor_load_to_lds(g0, g1, g2, g3, 0);
#endif
    __builtin_amdgcn_s_wait_tensorcnt((unsigned short)0);
  }
  __syncthreads();
#else
  for (int i = threadIdx.x; i < 64 * DDIM; i += 256) tile[i] = LOC[base + i];
  __syncthreads();
#endif
  // convert LDS fp32 -> bf16, 8 elements per thread per step
  for (int i = threadIdx.x * 8; i < 64 * DDIM; i += 256 * 8) {
    v8bf v;
    #pragma unroll
    for (int j = 0; j < 8; ++j) v[j] = (__bf16)tile[i + j];
    *(v8bf*)(LOCp + base + i) = v;
  }
}

// ---------------- pack QKV into fragment-friendly bf16 ----------------
// Qb,Kb: [H][NN][32] (dh=16, upper half zero). Vt: [H][16][NNP] (transposed).
__global__ void pack_qkv_k(const float* __restrict__ qkv, __bf16* __restrict__ Qb,
                           __bf16* __restrict__ Kb, __bf16* __restrict__ Vt) {
  const int n = blockIdx.x;
  const int t = threadIdx.x;           // 64
  const int h = t >> 4, d = t & 15;
  const size_t rb = (size_t)n * (3 * DDIM) + h * 16 + d;
  const float q = qkv[rb];
  const float k = qkv[rb + DDIM];
  const float v = qkv[rb + 2 * DDIM];
  const size_t qi = ((size_t)h * NN + n) * 32 + d;
  Qb[qi] = (__bf16)q; Qb[qi + 16] = (__bf16)0.0f;
  Kb[qi] = (__bf16)k; Kb[qi + 16] = (__bf16)0.0f;
  Vt[((size_t)h * 16 + d) * NNP + n] = (__bf16)v;
}

// ---------------- fused two-pass attention ----------------
__global__ __launch_bounds__(32)
void attn_fused_k(const __bf16* __restrict__ Qb, const __bf16* __restrict__ Kb,
                  const __bf16* __restrict__ Vt, float* __restrict__ attn_mean,
                  float* __restrict__ attn_out) {
  const int lane = threadIdx.x;
  const int r    = lane & 15;
  const int hi   = lane >> 4;
  const int h    = blockIdx.y;
  const int m0   = blockIdx.x * 16;
  const float scale = 0.25f;           // 1/sqrt(16)

  __shared__ __attribute__((aligned(32))) __bf16 sP[16 * 32];

  const v16bf aQ = load_frag_split(Qb + ((size_t)h * NN + (m0 + r)) * 32 + hi * 16);
  const __bf16* kbase = Kb + (size_t)h * NN * 32;
  const __bf16* vbase = Vt + (size_t)h * 16 * NNP;

  float m[8], l[8];
  #pragma unroll
  for (int i = 0; i < 8; ++i) { m[i] = -1e30f; l[i] = 0.0f; }
  // zero P pad region once (upper K half; aQ pad makes B upper half don't-care, but P
  // is an A-operand so its pad columns must be true zeros)
  sP[r * 32 + 16 + hi * 8 + 0] = (__bf16)0.0f;
  #pragma unroll
  for (int i = 1; i < 8; ++i) sP[r * 32 + 16 + hi * 8 + i] = (__bf16)0.0f;

  // ---- pass 1: online row max / sum ----
  for (int j = 0; j < NN / 16; ++j) {
    const v16bf bK = *(const v16bf*)(kbase + (size_t)(j * 16 + r) * 32 + hi * 16);
    v8f z = {};
    const v8f s = wmma_bf16(aQ, bK, z);
    #pragma unroll
    for (int i = 0; i < 8; ++i) {
      const float sv = s[i] * scale;
      float tm = sv;
      #pragma unroll
      for (int d = 1; d < 16; d <<= 1) tm = fmaxf(tm, __shfl_xor(tm, d, 32));
      const float mn = fmaxf(m[i], tm);
      float ts = __expf(sv - mn);
      #pragma unroll
      for (int d = 1; d < 16; d <<= 1) ts += __shfl_xor(ts, d, 32);
      l[i] = l[i] * __expf(m[i] - mn) + ts;
      m[i] = mn;
    }
  }
  float linv[8];
  #pragma unroll
  for (int i = 0; i < 8; ++i) linv[i] = 1.0f / l[i];

  // ---- pass 2: probs -> attn_mean (atomic), O += P @ V ----
  v8f o = {};
  for (int j = 0; j < NN / 16; ++j) {
    const v16bf bK = *(const v16bf*)(kbase + (size_t)(j * 16 + r) * 32 + hi * 16);
    v8f z = {};
    const v8f s = wmma_bf16(aQ, bK, z);
    #pragma unroll
    for (int i = 0; i < 8; ++i) {
      const float p   = __expf(s[i] * scale - m[i]) * linv[i];
      const int   row = m0 + i + hi * 8;
      const int   col = j * 16 + r;
      atomicAdd(&attn_mean[(size_t)row * NN + col], p * (1.0f / (float)NHEADS));
      sP[(i + hi * 8) * 32 + r] = (__bf16)p;
    }
    __syncthreads();
    const v16bf aP = load_frag_split(sP + r * 32 + hi * 16);
    // V^T row r = output dim, keys contiguous; P pad cols are zero so the hi=1
    // half may read the next key tile harmlessly (Vt is padded by 16 at the end).
    const v16bf bV = *(const v16bf*)(vbase + (size_t)r * NNP + j * 16 + hi * 16);
    o = wmma_bf16(aP, bV, o);
    __syncthreads();
  }

  const int col = h * 16 + r;
  #pragma unroll
  for (int i = 0; i < 8; ++i)
    attn_out[(size_t)(m0 + i + hi * 8) * DDIM + col] = o[i];
}

// ---------------- argmin per row (negative sampling) ----------------
__global__ void argmin_row_k(const float* __restrict__ am, int* __restrict__ samp) {
  __shared__ float sv[256];
  __shared__ int   si[256];
  const int row = blockIdx.x, t = threadIdx.x;
  float best = 3.4e38f; int bi = 0;
  for (int c = t; c < NN; c += 256) {
    const float v = am[(size_t)row * NN + c];
    if (v < best) { best = v; bi = c; }
  }
  sv[t] = best; si[t] = bi;
  __syncthreads();
  for (int s = 128; s > 0; s >>= 1) {
    if (t < s && sv[t + s] < sv[t]) { sv[t] = sv[t + s]; si[t] = si[t + s]; }
    __syncthreads();
  }
  if (t == 0) samp[row] = si[0];
}

// ---------------- bilinear logits ----------------
__global__ void logits_k(const float* __restrict__ tmp, const float* __restrict__ glb,
                         const int* __restrict__ samp, const float* __restrict__ bil_b,
                         float* __restrict__ out_pos, float* __restrict__ out_neg) {
  __shared__ float sp[64], sn[64];
  const int n = blockIdx.x, t = threadIdx.x;
  const float a = tmp[(size_t)n * DDIM + t];
  sp[t] = a * glb[(size_t)n * DDIM + t];
  sn[t] = a * glb[(size_t)samp[n] * DDIM + t];
  __syncthreads();
  for (int s = 32; s > 0; s >>= 1) {
    if (t < s) { sp[t] += sp[t + s]; sn[t] += sn[t + s]; }
    __syncthreads();
  }
  if (t == 0) {
    out_pos[n] = sp[0] + bil_b[0];
    out_neg[n] = sn[0] + bil_b[0];
  }
}

// ---------------- host orchestration ----------------
static inline float* carve(float** p, size_t nfloats) {
  float* r = *p;
  *p += (nfloats + 15) & ~(size_t)15;   // keep 64B alignment
  return r;
}

extern "C" void kernel_launch(void* const* d_in, const int* in_sizes, int n_in,
                              void* d_out, int out_size, void* d_ws, size_t ws_size,
                              hipStream_t stream) {
  const float* x        = (const float*)d_in[0];
  const int*   edge     = (const int*)  d_in[1];
  const int*   sub      = (const int*)  d_in[2];
  const float* enc_w1   = (const float*)d_in[3];
  const float* enc_b1   = (const float*)d_in[4];
  const float* enc_w2   = (const float*)d_in[5];
  const float* enc_b2   = (const float*)d_in[6];
  const float* dec_w1   = (const float*)d_in[7];
  const float* dec_b1   = (const float*)d_in[8];
  const float* dec_w2   = (const float*)d_in[9];
  const float* dec_b2   = (const float*)d_in[10];
  const float* att_in_w = (const float*)d_in[11];
  const float* att_in_b = (const float*)d_in[12];
  const float* att_o_w  = (const float*)d_in[13];
  const float* att_o_b  = (const float*)d_in[14];
  const float* bil_w    = (const float*)d_in[15];
  const float* bil_b    = (const float*)d_in[16];

  const int* esrc = edge;
  const int* edst = edge + NE;

  float* recon      = (float*)d_out;
  float* logits_pos = recon + (size_t)NN * IN_DIM;
  float* logits_neg = logits_pos + NN;

  float* p = (float*)d_ws;
  float* inv  = carve(&p, NN);
  float* Hh   = carve(&p, (size_t)NN * H1DIM);
  float* AGG1 = carve(&p, (size_t)NN * H1DIM);
  float* H2   = carve(&p, (size_t)NN * DDIM);
  float* AGG2 = carve(&p, (size_t)NN * DDIM);
  float* LOC  = carve(&p, (size_t)NN * DDIM);
  float* DEC1 = carve(&p, (size_t)NN * H1DIM);
  float* QKV  = carve(&p, (size_t)NN * 3 * DDIM);
  float* AM   = carve(&p, (size_t)NN * NN);
  float* AO   = carve(&p, (size_t)NN * DDIM);
  float* GLB  = carve(&p, (size_t)NN * DDIM);
  float* TMPB = carve(&p, (size_t)NN * DDIM);
  int*   SAMP = (int*)carve(&p, NN);
  // bf16 packs (carved in float units = 2 bf16 each)
  __bf16* Xp    = (__bf16*)carve(&p, (size_t)NN * INP / 2);
  __bf16* W1t   = (__bf16*)carve(&p, (size_t)H1DIM * INP / 2);
  __bf16* H1p   = (__bf16*)carve(&p, (size_t)NN * H1DIM / 2);
  __bf16* W2t   = (__bf16*)carve(&p, (size_t)DDIM * H1DIM / 2);
  __bf16* LOCp  = (__bf16*)carve(&p, (size_t)NN * DDIM / 2);
  __bf16* DW1t  = (__bf16*)carve(&p, (size_t)H1DIM * DDIM / 2);
  __bf16* DEC1p = (__bf16*)carve(&p, (size_t)NN * H1DIM / 2);
  __bf16* DW2t  = (__bf16*)carve(&p, (size_t)INP * H1DIM / 2);
  __bf16* AIW   = (__bf16*)carve(&p, (size_t)(3 * DDIM) * DDIM / 2);
  __bf16* AOW   = (__bf16*)carve(&p, (size_t)DDIM * DDIM / 2);
  __bf16* AOp   = (__bf16*)carve(&p, (size_t)NN * DDIM / 2);
  __bf16* BILt  = (__bf16*)carve(&p, (size_t)DDIM * DDIM / 2);
  __bf16* Qb    = (__bf16*)carve(&p, (size_t)NHEADS * NN * 32 / 2);
  __bf16* Kb    = (__bf16*)carve(&p, (size_t)NHEADS * NN * 32 / 2);
  __bf16* Vt    = (__bf16*)carve(&p, (size_t)NHEADS * 16 * NNP / 2);

  // per-call zeroing of accumulators / padded buffers
  zero_k<<<64,   256, 0, stream>>>(inv,  (size_t)NN);
  zero_k<<<2048, 256, 0, stream>>>(AGG1, (size_t)NN * H1DIM);
  zero_k<<<512,  256, 0, stream>>>(AGG2, (size_t)NN * DDIM);
  zero_k<<<8192, 256, 0, stream>>>(AM,   (size_t)NN * NN);
  zero_k<<<256,  256, 0, stream>>>((float*)Vt, (size_t)NHEADS * 16 * NNP / 2);

  // pack weights + x (all reused straight out of L2)
  pack_a_k <<<2048, 256, 0, stream>>>(x,      Xp,   NN, IN_DIM, INP);
  pack_bt_k<<<1024, 256, 0, stream>>>(enc_w1, W1t,  IN_DIM, H1DIM, H1DIM, INP, 1);
  pack_bt_k<<<64,   256, 0, stream>>>(enc_w2, W2t,  H1DIM, DDIM, DDIM, H1DIM, 1);
  pack_bt_k<<<64,   256, 0, stream>>>(dec_w1, DW1t, DDIM, H1DIM, H1DIM, DDIM, 1);
  pack_bt_k<<<1024, 256, 0, stream>>>(dec_w2, DW2t, H1DIM, IN_DIM, INP, H1DIM, 1);
  pack_bt_k<<<16,   256, 0, stream>>>(att_in_w, AIW, DDIM, 3 * DDIM, 3 * DDIM, DDIM, 0);
  pack_bt_k<<<4,    256, 0, stream>>>(att_o_w,  AOW, DDIM, DDIM, DDIM, DDIM, 0);
  pack_bt_k<<<4,    256, 0, stream>>>(bil_w,    BILt, DDIM, DDIM, DDIM, DDIM, 1);

  // degrees -> D^-1/2
  deg_acc_k<<<192, 256, 0, stream>>>(edst, inv, NE);
  inv_sqrt_k<<<NN / 256, 256, 0, stream>>>(inv, NN);

  // ---- GCN conv 1 ----
  gemm_bf16_k<<<dim3(H1DIM / 16, NN / 16), 32, 0, stream>>>(
      Xp, W1t, nullptr, Hh, NN, H1DIM, INP, 0, 1.0f);
  edge_agg_k<<<NE, 256, 0, stream>>>(esrc, edst, Hh, inv, AGG1, NE, H1DIM);
  gcn_finalize_k<<<NN, 256, 0, stream>>>(AGG1, Hh, inv, enc_b1, H1DIM, 1);

  // ---- GCN conv 2 ----
  pack_a_k<<<1024, 256, 0, stream>>>(AGG1, H1p, NN, H1DIM, H1DIM);
  gemm_bf16_k<<<dim3(DDIM / 16, NN / 16), 32, 0, stream>>>(
      H1p, W2t, nullptr, H2, NN, DDIM, H1DIM, 0, 1.0f);
  edge_agg_k<<<NE, 64, 0, stream>>>(esrc, edst, H2, inv, AGG2, NE, DDIM);
  gcn_finalize_k<<<NN, 64, 0, stream>>>(AGG2, H2, inv, enc_b2, DDIM, 0);

  // ---- local niches ----
  gather_mean_k<<<NN, DDIM, 0, stream>>>(AGG2, sub, LOC);
  pack_loc_tdm_k<<<NN / 64, 256, 0, stream>>>(LOC, LOCp);   // TDM-staged pack

  // ---- decoder ----
  gemm_bf16_k<<<dim3(H1DIM / 16, NN / 16), 32, 0, stream>>>(
      LOCp, DW1t, dec_b1, DEC1, NN, H1DIM, DDIM, 1, 1.0f);
  pack_a_k<<<1024, 256, 0, stream>>>(DEC1, DEC1p, NN, H1DIM, H1DIM);
  gemm_bf16_k<<<dim3(INP / 16, NN / 16), 32, 0, stream>>>(
      DEC1p, DW2t, dec_b2, recon, NN, IN_DIM, H1DIM, 0, 1.0f);

  // ---- MHA ----
  gemm_bf16_k<<<dim3((3 * DDIM) / 16, NN / 16), 32, 0, stream>>>(
      LOCp, AIW, att_in_b, QKV, NN, 3 * DDIM, DDIM, 0, 1.0f);
  pack_qkv_k<<<NN, 64, 0, stream>>>(QKV, Qb, Kb, Vt);
  attn_fused_k<<<dim3(NN / 16, NHEADS), 32, 0, stream>>>(Qb, Kb, Vt, AM, AO);
  pack_a_k<<<256, 256, 0, stream>>>(AO, AOp, NN, DDIM, DDIM);
  gemm_bf16_k<<<dim3(DDIM / 16, NN / 16), 32, 0, stream>>>(
      AOp, AOW, att_o_b, GLB, NN, DDIM, DDIM, 0, 1.0f);

  // ---- negative sampling + bilinear discriminator ----
  argmin_row_k<<<NN, 256, 0, stream>>>(AM, SAMP);
  gemm_bf16_k<<<dim3(DDIM / 16, NN / 16), 32, 0, stream>>>(
      LOCp, BILt, nullptr, TMPB, NN, DDIM, DDIM, 0, 1.0f);
  logits_k<<<NN, 64, 0, stream>>>(TMPB, GLB, SAMP, bil_b, logits_pos, logits_neg);
}
